// _Attention_74663711474127
// MI455X (gfx1250) — matrix operand order repaired and checked by reading, not compiled, not run on hardware
//
#include <hip/hip_runtime.h>

// Problem constants (fixed by the reference setup).
#define B_    256
#define T_    64
#define I_    512
#define H_    512
#define C_    97
#define STEPS 26          // max_length + 1
#define G4H   2048        // 4*H
#define KX    1024        // I + H : fused [ctx|h] GEMM K
#define NCPAD 128         // C padded so every wave's 64-wide supertile is valid
#define XDIM  609         // I + C  (W_ih row stride)

typedef __bf16 bf16x16 __attribute__((ext_vector_type(16)));
typedef float  f32x8   __attribute__((ext_vector_type(8)));
typedef unsigned int u32x4 __attribute__((ext_vector_type(4)));
typedef int          i32x4 __attribute__((ext_vector_type(4)));
typedef int          i32x8 __attribute__((ext_vector_type(8)));

union Frag16 { bf16x16 v; unsigned int u[8]; uint4 q[2]; };

static __device__ __forceinline__ unsigned short f32_to_bf16(float f) {
  unsigned int u = __float_as_uint(f);
  u = (u + 0x7FFFu + ((u >> 16) & 1u)) >> 16;   // round-to-nearest-even
  return (unsigned short)u;
}

#if __has_builtin(__builtin_amdgcn_tensor_load_to_lds) && \
    __has_builtin(__builtin_amdgcn_s_wait_tensorcnt)
#define USE_TDM 1
#else
#define USE_TDM 0
#endif

#if USE_TDM
// -----------------------------------------------------------------------------
// TDM: DMA one 16x32 bf16 tile (row stride = stride_elems) from global to LDS.
// D# per CDNA5 ISA ch.8: group0 = {count/flags, lds_addr, global_addr, type=2},
// group1 = {data_size=2B, tensor_dim0/1, tile_dim0=32/tile_dim1=16, dim0 stride}.
// -----------------------------------------------------------------------------
static __device__ __forceinline__ void tdm_load_a(const unsigned short* gsrc,
                                                  unsigned lds_off,
                                                  unsigned stride_elems,
                                                  unsigned rem_elems)
{
  unsigned long long ga = (unsigned long long)gsrc;
  u32x4 g0;
  g0[0] = 1u;                                             // count=1, no gather
  g0[1] = lds_off;                                        // lds_addr (bytes)
  g0[2] = (unsigned)(ga & 0xffffffffu);                   // global_addr lo
  g0[3] = (unsigned)((ga >> 32) & 0x01ffffffu) | (2u << 30); // addr hi | type=2
  i32x8 g1;
  g1[0] = (int)(1u << 16);                                // data_size=1 -> 2 bytes
  g1[1] = (int)((rem_elems & 0xffffu) << 16);             // tensor_dim0[15:0]
  g1[2] = (int)(((rem_elems >> 16) & 0xffffu) | (16u << 16)); // dim0 hi | tensor_dim1=16
  g1[3] = (int)(32u << 16);                               // dim1 hi=0 | tile_dim0=32
  g1[4] = (int)16u;                                       // tile_dim1=16, tile_dim2=0
  g1[5] = (int)stride_elems;                              // tensor_dim0_stride lo
  g1[6] = 0;                                              // stride hi | dim1_stride lo
  g1[7] = 0;                                              // dim1_stride hi
  i32x4 z4 = {0, 0, 0, 0};
#if __clang_major__ >= 23
  i32x8 z8 = {0, 0, 0, 0, 0, 0, 0, 0};
  __builtin_amdgcn_tensor_load_to_lds(g0, g1, z4, z4, z8, 0);
#else
  __builtin_amdgcn_tensor_load_to_lds(g0, g1, z4, z4, 0);
#endif
}
#endif

// Cooperative fallback stage: copy 16x32 bf16 tile (1KB) to LDS.
static __device__ __forceinline__ void stage_copy(const unsigned short* __restrict__ src,
                                                  unsigned short* __restrict__ dst,
                                                  int lda, int nthreads, int tid)
{
  for (int i = tid; i < 128; i += nthreads) {     // 128 x uint2 (8B) = 1KB
    const int e   = i << 2;                       // element index
    const int row = e >> 5, col = e & 31;
    *(uint2*)(dst + e) = *(const uint2*)(src + (size_t)row * lda + col);
  }
}

// ---------------------------------------------------------------------------
// bf16 WMMA GEMM: C[M,N] = A[M,K]*B[N,K]^T (+bias). Each wave owns a 16x64
// tile (4 accumulators); the block's shared 16-row A tile is staged into LDS
// by the Tensor Data Mover (double buffered), B streams from global/L2.
// Grid: x = M/16, y covers N: gridDim.y * (blockDim/32) * 64 == N (padded).
// ---------------------------------------------------------------------------
__global__ void __launch_bounds__(128)
wmma_gemm_bf16(const unsigned short* __restrict__ A, int lda,
               const unsigned short* __restrict__ Bm, int ldb,
               float* __restrict__ Cc, int ldc,
               const float* __restrict__ bias,
               int K, int nLimit)
{
  __shared__ __align__(16) unsigned short sA[2][16 * 32];

  const int tid   = threadIdx.x;
  const int lane  = tid & 31;
  const int wave  = tid >> 5;
  const int nwav  = blockDim.x >> 5;
  const int tileM = blockIdx.x;
  const int super = blockIdx.y * nwav + wave;          // 64-wide N supertile

  const int r16  = lane & 15;
  const int hi   = lane >> 4;
  const int kb16 = hi << 4;

  const unsigned short* arow0 = A + (size_t)tileM * 16 * lda;
  const unsigned short* brow0 = Bm + (size_t)(super * 64 + r16) * ldb;
  const unsigned short* brow1 = brow0 + (size_t)16 * ldb;
  const unsigned short* brow2 = brow0 + (size_t)32 * ldb;
  const unsigned short* brow3 = brow0 + (size_t)48 * ldb;

  f32x8 acc0 = {}, acc1 = {}, acc2 = {}, acc3 = {};

  // ---- stage K-chunk 0 into sA[0] ----
#if USE_TDM
  if (wave == 0)
    tdm_load_a(arow0, (unsigned)(size_t)(void*)&sA[0][0], (unsigned)lda, (unsigned)K);
  if (wave == 0) __builtin_amdgcn_s_wait_tensorcnt(0);
#else
  stage_copy(arow0, &sA[0][0], lda, blockDim.x, tid);
#endif
  __syncthreads();

  for (int k0 = 0; k0 < K; k0 += 32) {
    const int cur = (k0 >> 5) & 1;
    const bool more = (k0 + 32) < K;

    // ---- async stage next A chunk while computing this one ----
    if (more) {
#if USE_TDM
      if (wave == 0)
        tdm_load_a(arow0 + (k0 + 32), (unsigned)(size_t)(void*)&sA[cur ^ 1][0],
                   (unsigned)lda, (unsigned)(K - k0 - 32));
#else
      stage_copy(arow0 + (k0 + 32), &sA[cur ^ 1][0], lda, blockDim.x, tid);
#endif
      __builtin_prefetch(brow0 + k0 + 32, 0, 0);     // gfx1250 global_prefetch_b8
      __builtin_prefetch(brow2 + k0 + 32, 0, 0);
    }

    // ---- A fragment from LDS (contiguous k = 16*hi + s, matching B) ----
    Frag16 a;
    const unsigned short* sa = &sA[cur][0] + (r16 << 5) + (hi << 4);
    a.q[0] = *(const uint4*)(sa);
    a.q[1] = *(const uint4*)(sa + 8);

    // ---- 4 B fragments (K pairs contiguous in the [N][K] weight rows) ----
    Frag16 b0, b1, b2, b3;
    b0.q[0] = *(const uint4*)(brow0 + k0 + kb16);
    b0.q[1] = *(const uint4*)(brow0 + k0 + kb16 + 8);
    b1.q[0] = *(const uint4*)(brow1 + k0 + kb16);
    b1.q[1] = *(const uint4*)(brow1 + k0 + kb16 + 8);
    b2.q[0] = *(const uint4*)(brow2 + k0 + kb16);
    b2.q[1] = *(const uint4*)(brow2 + k0 + kb16 + 8);
    b3.q[0] = *(const uint4*)(brow3 + k0 + kb16);
    b3.q[1] = *(const uint4*)(brow3 + k0 + kb16 + 8);

    acc0 = __builtin_amdgcn_wmma_f32_16x16x32_bf16(false, a.v, false, b0.v, (short)0, acc0, false, false);
    acc1 = __builtin_amdgcn_wmma_f32_16x16x32_bf16(false, a.v, false, b1.v, (short)0, acc1, false, false);
    acc2 = __builtin_amdgcn_wmma_f32_16x16x32_bf16(false, a.v, false, b2.v, (short)0, acc2, false, false);
    acc3 = __builtin_amdgcn_wmma_f32_16x16x32_bf16(false, a.v, false, b3.v, (short)0, acc3, false, false);

    if (more) {
#if USE_TDM
      if (wave == 0) __builtin_amdgcn_s_wait_tensorcnt(0);
#endif
      __syncthreads();
    }
  }

  // ---- store: C layout lanes 0-15 -> M=tile+r, lanes 16-31 -> M=tile+8+r ----
  const int mbase = tileM * 16 + (hi << 3);
  const int nbase = super * 64 + r16;
#pragma unroll
  for (int s = 0; s < 4; ++s) {
    const int ncol = nbase + s * 16;
    if (ncol < nLimit) {
      const float bs = bias ? bias[ncol] : 0.0f;
      float* cp = Cc + (size_t)mbase * ldc + ncol;
      const f32x8 acc = (s == 0) ? acc0 : (s == 1) ? acc1 : (s == 2) ? acc2 : acc3;
#pragma unroll
      for (int r = 0; r < 8; ++r)
        cp[(size_t)r * ldc] = acc[r] + bs;
    }
  }
}

// ---------------------------------------------------------------------------
// Prep kernels
// ---------------------------------------------------------------------------
__global__ void cvt_f32_bf16(const float* __restrict__ in,
                             unsigned short* __restrict__ out, int n) {
  int i = blockIdx.x * blockDim.x + threadIdx.x;
  if (i < n) out[i] = f32_to_bf16(in[i]);
}

// Wx[j, 0:512] = W_ih[j, 0:512]; Wx[j, 512:1024] = W_hh[j, :]
__global__ void build_wx(const float* __restrict__ W_ih,
                         const float* __restrict__ W_hh,
                         unsigned short* __restrict__ out) {
  int i = blockIdx.x * blockDim.x + threadIdx.x;   // G4H*KX threads
  int j = i >> 10, k = i & 1023;
  float v = (k < I_) ? W_ih[(size_t)j * XDIM + k]
                     : W_hh[(size_t)j * H_ + (k - I_)];
  out[i] = f32_to_bf16(v);
}

// W_cls padded from [97,512] to [128,512] bf16 (zero pad rows)
__global__ void build_wcls(const float* __restrict__ W_cls,
                           unsigned short* __restrict__ out) {
  int i = blockIdx.x * blockDim.x + threadIdx.x;   // NCPAD*H_ threads
  int row = i >> 9, k = i & 511;
  float v = (row < C_) ? W_cls[(size_t)row * H_ + k] : 0.0f;
  out[i] = f32_to_bf16(v);
}

__global__ void build_bias2(const float* __restrict__ a,
                            const float* __restrict__ b,
                            float* __restrict__ out) {
  int i = blockIdx.x * blockDim.x + threadIdx.x;
  if (i < G4H) out[i] = a[i] + b[i];
}

// ---------------------------------------------------------------------------
// Attention: per block b: score[t] = sum_h tanh(proj[b,t,h]+pp[b,h])*sv[h]
// softmax over T, ctx[i] = sum_t attn[t]*bh[b,t,i] -> bf16 into xh[b, 0:512]
// ---------------------------------------------------------------------------
__global__ void __launch_bounds__(256)
attn_kernel(const float* __restrict__ proj, const float* __restrict__ pp,
            const float* __restrict__ sv,   const float* __restrict__ bh,
            unsigned short* __restrict__ xh)
{
  const int b    = blockIdx.x;
  const int tid  = threadIdx.x;
  const int lane = tid & 31;
  const int wave = tid >> 5;

  __shared__ float s_attn[T_];
  __shared__ float s_inv;

  const float* projb = proj + (size_t)b * T_ * H_;
  const float* ppb   = pp   + (size_t)b * H_;

  for (int t = wave; t < T_; t += 8) {
    const float* pr = projb + (size_t)t * H_;
    float s = 0.0f;
    for (int h = lane; h < H_; h += 32)
      s += tanhf(pr[h] + ppb[h]) * sv[h];
    s += __shfl_xor(s, 16);
    s += __shfl_xor(s, 8);
    s += __shfl_xor(s, 4);
    s += __shfl_xor(s, 2);
    s += __shfl_xor(s, 1);
    if (lane == 0) s_attn[t] = s;
  }
  __syncthreads();

  if (tid == 0) {
    float mx = s_attn[0];
    for (int t = 1; t < T_; ++t) mx = fmaxf(mx, s_attn[t]);
    float sum = 0.0f;
    for (int t = 0; t < T_; ++t) { float e = expf(s_attn[t] - mx); s_attn[t] = e; sum += e; }
    s_inv = 1.0f / sum;
  }
  __syncthreads();

  const float inv = s_inv;
  const float* bhb = bh + (size_t)b * T_ * I_;
  for (int i = tid; i < I_; i += 256) {
    float acc = 0.0f;
#pragma unroll 4
    for (int t = 0; t < T_; ++t)
      acc += s_attn[t] * bhb[(size_t)t * I_ + i];
    xh[(size_t)b * KX + i] = f32_to_bf16(acc * inv);
  }
}

// ---------------------------------------------------------------------------
// LSTM pointwise: add one-hot column gather of W_ih, gate nonlinearities,
// update c, write h (bf16) into xh[:,512:] for the next step and into hoB.
// ---------------------------------------------------------------------------
__global__ void __launch_bounds__(256)
lstm_kernel(const float* __restrict__ gates, const float* __restrict__ W_ih,
            const int* __restrict__ ctxi, float* __restrict__ cst,
            unsigned short* __restrict__ xh, unsigned short* __restrict__ hoB,
            int step)
{
  const int b    = blockIdx.x;
  const int hidx = blockIdx.y * blockDim.x + threadIdx.x;   // 0..H_-1
  const int cv   = ctxi[b * STEPS + step];                  // context[b, step]
  const float* g = gates + (size_t)b * G4H;

  const size_t oh = (size_t)I_ + cv;                        // one-hot column
  float ig = g[hidx]          + W_ih[(size_t)(hidx)          * XDIM + oh];
  float fg = g[H_ + hidx]     + W_ih[(size_t)(H_ + hidx)     * XDIM + oh];
  float gg = g[2 * H_ + hidx] + W_ih[(size_t)(2 * H_ + hidx) * XDIM + oh];
  float og = g[3 * H_ + hidx] + W_ih[(size_t)(3 * H_ + hidx) * XDIM + oh];

  const float si = 1.0f / (1.0f + expf(-ig));
  const float sf = 1.0f / (1.0f + expf(-fg));
  const float so = 1.0f / (1.0f + expf(-og));

  const size_t ci = (size_t)b * H_ + hidx;
  const float c_new = sf * cst[ci] + si * tanhf(gg);
  const float h_new = so * tanhf(c_new);
  cst[ci] = c_new;

  const unsigned short hb = f32_to_bf16(h_new);
  xh[(size_t)b * KX + I_ + hidx] = hb;
  hoB[((size_t)b * STEPS + step) * H_ + hidx] = hb;
}

// ---------------------------------------------------------------------------
extern "C" void kernel_launch(void* const* d_in, const int* in_sizes, int n_in,
                              void* d_out, int out_size, void* d_ws, size_t ws_size,
                              hipStream_t stream)
{
  (void)in_sizes; (void)n_in; (void)out_size; (void)ws_size;

  const float* bh     = (const float*)d_in[0];
  const int*   ctxi   = (const int*)d_in[1];
  const float* W_i2h  = (const float*)d_in[4];
  const float* W_h2h  = (const float*)d_in[5];
  const float* b_h2h  = (const float*)d_in[6];
  const float* wscore = (const float*)d_in[7];   // [1,H] -> [H]
  const float* W_ih   = (const float*)d_in[8];
  const float* b_ih   = (const float*)d_in[9];
  const float* W_hh   = (const float*)d_in[10];
  const float* b_hh   = (const float*)d_in[11];
  const float* W_cls  = (const float*)d_in[12];
  const float* b_cls  = (const float*)d_in[13];
  float* out = (float*)d_out;                    // [B, STEPS, C]

  char* wsp = (char*)d_ws;
  auto take = [&](size_t bytes) {
    char* p = wsp; wsp += (bytes + 255) & ~(size_t)255; return p;
  };
  unsigned short* bhB   = (unsigned short*)take((size_t)B_ * T_ * I_ * 2);   // bf16 batch_hidden
  unsigned short* WiB   = (unsigned short*)take((size_t)H_ * I_ * 2);        // bf16 W_i2h
  unsigned short* WhB   = (unsigned short*)take((size_t)H_ * H_ * 2);        // bf16 W_h2h
  unsigned short* WxB   = (unsigned short*)take((size_t)G4H * KX * 2);       // bf16 [W_ih[:,:I] | W_hh]
  unsigned short* WcB   = (unsigned short*)take((size_t)NCPAD * H_ * 2);     // bf16 W_cls padded
  float*          bias2 = (float*)take((size_t)G4H * 4);                     // b_ih + b_hh
  float*          proj  = (float*)take((size_t)B_ * T_ * H_ * 4);
  float*          pp    = (float*)take((size_t)B_ * H_ * 4);
  float*          gates = (float*)take((size_t)B_ * G4H * 4);
  unsigned short* xh    = (unsigned short*)take((size_t)B_ * KX * 2);        // bf16 [ctx | h]
  float*          cst   = (float*)take((size_t)B_ * H_ * 4);
  unsigned short* hoB   = (unsigned short*)take((size_t)B_ * STEPS * H_ * 2);

  // ---- prep: conversions + fused weights + zero state ----
  {
    int n = B_ * T_ * I_;
    cvt_f32_bf16<<<(n + 255) / 256, 256, 0, stream>>>(bh, bhB, n);
    n = H_ * I_;
    cvt_f32_bf16<<<(n + 255) / 256, 256, 0, stream>>>(W_i2h, WiB, n);
    n = H_ * H_;
    cvt_f32_bf16<<<(n + 255) / 256, 256, 0, stream>>>(W_h2h, WhB, n);
    n = G4H * KX;
    build_wx<<<(n + 255) / 256, 256, 0, stream>>>(W_ih, W_hh, WxB);
    n = NCPAD * H_;
    build_wcls<<<(n + 255) / 256, 256, 0, stream>>>(W_cls, WcB);
    build_bias2<<<(G4H + 255) / 256, 256, 0, stream>>>(b_ih, b_hh, bias2);
    hipMemsetAsync(xh, 0, (size_t)B_ * KX * 2, stream);       // h0 = 0
    hipMemsetAsync(cst, 0, (size_t)B_ * H_ * 4, stream);      // c0 = 0
  }

  // ---- proj = batch_hidden @ W_i2h^T : M=16384, N=512, K=512 ----
  // N coverage: gridDim.y(2) * 4 waves * 64 = 512
  wmma_gemm_bf16<<<dim3((B_ * T_) / 16, 2), 128, 0, stream>>>(
      bhB, I_, WiB, I_, proj, H_, nullptr, I_, H_);

  // ---- recurrence ----
  for (int s = 0; s < STEPS; ++s) {
    // prev_proj = h @ W_h2h^T + b_h2h : A = xh[:,512:], M=256,N=512,K=512
    wmma_gemm_bf16<<<dim3(B_ / 16, 2), 128, 0, stream>>>(
        xh + I_, KX, WhB, H_, pp, H_, b_h2h, H_, H_);

    // attention -> ctx (bf16) into xh[:, :512]
    attn_kernel<<<B_, 256, 0, stream>>>(proj, pp, wscore, bh, xh);

    // gates = [ctx|h] @ [W_ih[:,:512]|W_hh]^T + (b_ih+b_hh) : M=256,N=2048,K=1024
    wmma_gemm_bf16<<<dim3(B_ / 16, 8), 128, 0, stream>>>(
        xh, KX, WxB, KX, gates, G4H, bias2, KX, G4H);

    // pointwise LSTM + one-hot column gather; writes h (bf16) and c
    lstm_kernel<<<dim3(B_, H_ / 256), 256, 0, stream>>>(
        gates, W_ih, ctxi, cst, xh, hoB, s);
  }

  // ---- probs = hidden_outputs @ W_cls^T + b_cls : M=6656, N=128(97), K=512 ----
  // block = 64 threads (2 waves): 1 * 2 * 64 = 128 = NCPAD
  wmma_gemm_bf16<<<dim3((B_ * STEPS) / 16, 1), 64, 0, stream>>>(
      hoB, H_, WcB, H_, out, C_, b_cls, H_, C_);
}
